// MultiHeadAttention_37580963840030
// MI455X (gfx1250) — compile-verified, matrix operand order checked by reading, not run on hardware
//
#include <hip/hip_runtime.h>

// ---------------------------------------------------------------------------
// MHA on MI455X (gfx1250): all matmuls on v_wmma_f32_16x16x32_bf16 (wave32).
// Staging via GLOBAL_LOAD_ASYNC_TO_LDS_B128 (ASYNCcnt) with double-buffered
// LDS; softmax max-reduction via DPP16 row_xmask; softmax sum kept as
// per-lane partials and reduced once in the epilogue.
// Pipeline: cvt fp32->bf16  ->  GEMM qkv  ->  flash-attention  ->  GEMM proj.
// ---------------------------------------------------------------------------

typedef __bf16 bf16_t;
typedef __bf16 v16bf __attribute__((ext_vector_type(16)));
typedef float  v8f   __attribute__((ext_vector_type(8)));

#define DIMC   2048
#define NHEAD  16
#define DH     128
#define BATCH  4
#define SEQ    2048
#define C3     (3 * DIMC)       // 6144
#define MTOT   (BATCH * SEQ)    // 8192

// Async copy 16B per lane: global -> LDS, offset applied to both sides; tracked by ASYNCcnt.
#define ASYNC_B128(ldsoff, gaddr, imm)                                       \
    asm volatile("global_load_async_to_lds_b128 %0, %1, off offset:" #imm    \
                 :: "v"(ldsoff), "v"(gaddr) : "memory")
#define WAIT_ASYNC0() asm volatile("s_wait_asynccnt 0x0" ::: "memory")

__device__ __forceinline__ unsigned lds_lo32(const void* p) {
    return (unsigned)(unsigned long long)p;   // LDS aperture: offset in [31:0]
}

__device__ __forceinline__ v8f wmma_bf16(v16bf a, v16bf b, v8f c) {
    // 8 args: (neg_a, A, neg_b, B, c_mod, C, reuse_a, reuse_b)
    return __builtin_amdgcn_wmma_f32_16x16x32_bf16(false, a, false, b,
                                                   (short)0, c, false, false);
}

// K packing for 16-bit A/B fragments (ISA 7.12.2): VGPR v holds K pair
//   kbase = (v<4 ? 0 : 16) + (lane>=16 ? 8 : 0) + 2*(v&3)
__device__ __forceinline__ int kpack(int v, int hi) {
    return ((v < 4) ? 0 : 16) + hi * 8 + (v & 3) * 2;
}

// ---- DPP16 row_xmask butterfly over the 16-lane row (pure VALU) -----------
template<int CTRL>
__device__ __forceinline__ float dppmov(float x) {
    return __int_as_float(__builtin_amdgcn_update_dpp(
        0, __float_as_int(x), CTRL, 0xf, 0xf, true));
}
__device__ __forceinline__ float rowmax16(float x) {
    x = fmaxf(x, dppmov<0x168>(x));   // row_xmask:8
    x = fmaxf(x, dppmov<0x164>(x));   // row_xmask:4
    x = fmaxf(x, dppmov<0x162>(x));   // row_xmask:2
    x = fmaxf(x, dppmov<0x161>(x));   // row_xmask:1
    return x;
}
__device__ __forceinline__ float rowsum16(float x) {
    x += dppmov<0x168>(x);
    x += dppmov<0x164>(x);
    x += dppmov<0x162>(x);
    x += dppmov<0x161>(x);
    return x;
}

// --------------------------- fp32 -> bf16 ----------------------------------
__global__ __launch_bounds__(256)
void cvt_f32_bf16(const float* __restrict__ in, bf16_t* __restrict__ out) {
    int i = (blockIdx.x * 256 + threadIdx.x) * 4;
    float4 v = *(const float4*)(in + i);
    bf16_t o4[4] = {(bf16_t)v.x, (bf16_t)v.y, (bf16_t)v.z, (bf16_t)v.w};
    *(uint2*)(out + i) = *(const uint2*)o4;
}

// --------------------------- bf16 WMMA GEMM --------------------------------
// C[M,N] = A[M,K](bf16, row-major) * B[K,N](bf16, row-major) + bias
// Block: 128x128 tile, BK=64, 256 threads = 8 waves in 2x4; each wave 64x32.
// Double-buffered LDS filled by async-to-LDS; one barrier per K-stage.
template<bool OUT_BF16>
__global__ __launch_bounds__(256)
void gemm_bf16_wmma(const bf16_t* __restrict__ A, const bf16_t* __restrict__ Bm,
                    const float* __restrict__ bias, void* __restrict__ Out,
                    int M, int N, int K)
{
    __shared__ bf16_t ldsA[2][128][72];    // 128x64 tile, +8 pad
    __shared__ bf16_t ldsB[2][64][136];    // 64x128 tile, +8 pad

    const int tid  = threadIdx.x;
    const int lane = tid & 31;
    const int w    = tid >> 5;      // 0..7
    const int wr   = w >> 2;        // 0..1 (wave row)
    const int wc   = w & 3;         // 0..3 (wave col)
    const int r    = lane & 15;
    const int hi   = lane >> 4;

    const int m0 = blockIdx.x * 128;
    const int n0 = blockIdx.y * 128;

    v8f acc[4][2] = {};

    // staging assignment: A 128 rows x 2 segs of 32; B 64 rows x 4 segs of 32
    const int arow = tid >> 1, aseg = (tid & 1) * 32;
    const int brow = tid >> 2, bseg = (tid & 3) * 32;
    const bf16_t* gA = A + (size_t)(m0 + arow) * K + aseg;
    const bf16_t* gB = Bm + (size_t)brow * N + n0 + bseg;

    // ---- prologue: stage k0 = 0 into buffer 0 ----
    {
        unsigned la = lds_lo32(&ldsA[0][arow][aseg]);
        unsigned lb = lds_lo32(&ldsB[0][brow][bseg]);
        ASYNC_B128(la, gA, 0);  ASYNC_B128(la, gA, 16);
        ASYNC_B128(la, gA, 32); ASYNC_B128(la, gA, 48);
        ASYNC_B128(lb, gB, 0);  ASYNC_B128(lb, gB, 16);
        ASYNC_B128(lb, gB, 32); ASYNC_B128(lb, gB, 48);
    }
    WAIT_ASYNC0();
    __syncthreads();

    int cur = 0;
    for (int k0 = 0; k0 < K; k0 += 64) {
        // ---- issue async stage of the next K-slab into the other buffer ----
        if (k0 + 64 < K) {
            const bf16_t* nA = gA + (k0 + 64);
            const bf16_t* nB = gB + (size_t)(k0 + 64) * N;
            unsigned la = lds_lo32(&ldsA[cur ^ 1][arow][aseg]);
            unsigned lb = lds_lo32(&ldsB[cur ^ 1][brow][bseg]);
            ASYNC_B128(la, nA, 0);  ASYNC_B128(la, nA, 16);
            ASYNC_B128(la, nA, 32); ASYNC_B128(la, nA, 48);
            ASYNC_B128(lb, nB, 0);  ASYNC_B128(lb, nB, 16);
            ASYNC_B128(lb, nB, 32); ASYNC_B128(lb, nB, 48);
        }

        // ---- compute on current buffer: 2 sub-steps of K=32 ----
        #pragma unroll
        for (int ks = 0; ks < 64; ks += 32) {
            v16bf afr[4], bfr[2];
            #pragma unroll
            for (int mt = 0; mt < 4; ++mt) {
                const int rowi = wr * 64 + mt * 16 + r;
                #pragma unroll
                for (int v = 0; v < 8; ++v) {
                    const int kb = ks + kpack(v, hi);
                    afr[mt][2*v]   = ldsA[cur][rowi][kb];
                    afr[mt][2*v+1] = ldsA[cur][rowi][kb + 1];
                }
            }
            #pragma unroll
            for (int nt = 0; nt < 2; ++nt) {
                const int coli = wc * 32 + nt * 16 + r;
                #pragma unroll
                for (int v = 0; v < 8; ++v) {
                    const int kb = ks + kpack(v, hi);
                    bfr[nt][2*v]   = ldsB[cur][kb][coli];
                    bfr[nt][2*v+1] = ldsB[cur][kb + 1][coli];
                }
            }
            #pragma unroll
            for (int mt = 0; mt < 4; ++mt)
                #pragma unroll
                for (int nt = 0; nt < 2; ++nt)
                    acc[mt][nt] = wmma_bf16(afr[mt], bfr[nt], acc[mt][nt]);
        }

        // next buffer complete + everyone done reading current buffer
        WAIT_ASYNC0();
        __syncthreads();
        cur ^= 1;
    }

    // ---- epilogue: C layout (lane = col r; vgpr i = row i + 8*hi) ----
    #pragma unroll
    for (int mt = 0; mt < 4; ++mt) {
        const int gm = m0 + wr * 64 + mt * 16 + hi * 8;
        #pragma unroll
        for (int nt = 0; nt < 2; ++nt) {
            const int gn = n0 + wc * 32 + nt * 16 + r;
            const float bv = bias[gn];
            #pragma unroll
            for (int i = 0; i < 8; ++i) {
                const float val = acc[mt][nt][i] + bv;
                if (OUT_BF16)
                    ((bf16_t*)Out)[(size_t)(gm + i) * N + gn] = (bf16_t)val;
                else
                    ((float*)Out)[(size_t)(gm + i) * N + gn] = val;
            }
        }
    }
}

// --------------------------- flash attention -------------------------------
// Block per (b, h, 128 q rows); 8 waves x 16 q rows. Online softmax over
// key blocks of 32 (2 n-tiles). S and P*V both on WMMA bf16. K/V tiles
// double-buffered via async-to-LDS; max reduction via DPP row_xmask;
// the softmax sum stays per-lane and is reduced once in the epilogue.
__global__ __launch_bounds__(256)
void attn_wmma(const bf16_t* __restrict__ qkv, bf16_t* __restrict__ y)
{
    __shared__ bf16_t ldsK[2][32][136];
    __shared__ bf16_t ldsV[2][32][136];
    __shared__ bf16_t ldsP[8][16][40];

    const int tid  = threadIdx.x;
    const int lane = tid & 31;
    const int w    = tid >> 5;
    const int r    = lane & 15;
    const int hi   = lane >> 4;

    const int bid = blockIdx.x;          // 1024 = 16 qblk * 16 heads * 4 batch
    const int qb  = bid & 15;
    const int h   = (bid >> 4) & 15;
    const int b   = bid >> 8;

    const int q0 = qb * 128 + w * 16;    // wave's first q row
    const float scale = 0.08838834764831845f;   // 1/sqrt(128)

    // q fragments: A-layout 16x128 (4 K-chunks of 32), straight from global
    v16bf qf[4];
    const size_t qbase = ((size_t)b * SEQ + q0 + r) * C3 + (size_t)h * DH;
    #pragma unroll
    for (int c = 0; c < 4; ++c)
        #pragma unroll
        for (int v = 0; v < 8; ++v) {
            const int kb = kpack(v, hi);
            qf[c][2*v]   = qkv[qbase + c * 32 + kb];
            qf[c][2*v+1] = qkv[qbase + c * 32 + kb + 1];
        }

    v8f o[8] = {};
    float rowm[8], rowl[8];   // rowl: per-lane partial softmax denominator
    #pragma unroll
    for (int i = 0; i < 8; ++i) { rowm[i] = -1e30f; rowl[i] = 0.f; }

    // staging: K/V 32x128 tiles; each thread one 16-half row segment (2xb128)
    const int krow = tid >> 3, kseg = (tid & 7) * 16;
    const bf16_t* gKV = qkv + ((size_t)b * SEQ + krow) * C3
                            + DIMC + (size_t)h * DH + kseg;

    {   // prologue: stage key-block 0 into buffer 0
        unsigned lk = lds_lo32(&ldsK[0][krow][kseg]);
        unsigned lv = lds_lo32(&ldsV[0][krow][kseg]);
        ASYNC_B128(lk, gKV, 0);          ASYNC_B128(lk, gKV, 16);
        ASYNC_B128(lv, gKV + DIMC, 0);   ASYNC_B128(lv, gKV + DIMC, 16);
    }
    WAIT_ASYNC0();
    __syncthreads();

    int cur = 0;
    for (int kb0 = 0; kb0 < SEQ; kb0 += 32) {
        if (kb0 + 32 < SEQ) {   // async stage next key block
            const bf16_t* nKV = gKV + (size_t)(kb0 + 32) * C3;
            unsigned lk = lds_lo32(&ldsK[cur ^ 1][krow][kseg]);
            unsigned lv = lds_lo32(&ldsV[cur ^ 1][krow][kseg]);
            ASYNC_B128(lk, nKV, 0);          ASYNC_B128(lk, nKV, 16);
            ASYNC_B128(lv, nKV + DIMC, 0);   ASYNC_B128(lv, nKV + DIMC, 16);
        }

        // ---- S = (q K^T) * scale : two 16x16 n-tiles ----
        v8f s0 = {}, s1 = {};
        #pragma unroll
        for (int c = 0; c < 4; ++c) {
            v16bf bk0, bk1;
            #pragma unroll
            for (int v = 0; v < 8; ++v) {
                const int kb = kpack(v, hi);
                bk0[2*v]   = ldsK[cur][r][c*32 + kb];
                bk0[2*v+1] = ldsK[cur][r][c*32 + kb + 1];
                bk1[2*v]   = ldsK[cur][16 + r][c*32 + kb];
                bk1[2*v+1] = ldsK[cur][16 + r][c*32 + kb + 1];
            }
            s0 = wmma_bf16(qf[c], bk0, s0);
            s1 = wmma_bf16(qf[c], bk1, s1);
        }

        // ---- online softmax: row m = i + 8*hi lives in 16 lanes of a half.
        // corr is row-uniform (derived from the reduced max), so the
        // denominator partial sums can stay per-lane until the epilogue.
        #pragma unroll
        for (int i = 0; i < 8; ++i) {
            float a0 = s0[i] * scale, a1 = s1[i] * scale;
            const float mx   = rowmax16(fmaxf(a0, a1));
            const float nm   = fmaxf(rowm[i], mx);
            const float corr = __expf(rowm[i] - nm);
            rowm[i] = nm;
            const float p0 = __expf(a0 - nm), p1 = __expf(a1 - nm);
            rowl[i] = rowl[i] * corr + (p0 + p1);
            #pragma unroll
            for (int j = 0; j < 8; ++j) o[j][i] *= corr;
            // C-layout -> LDS so it can be reloaded in A-layout
            ldsP[w][hi * 8 + i][r]      = (bf16_t)p0;
            ldsP[w][hi * 8 + i][16 + r] = (bf16_t)p1;
        }

        // ---- O += P V  (K-dim = 32 keys) ----
        v16bf ap;
        #pragma unroll
        for (int v = 0; v < 8; ++v) {
            const int kb = kpack(v, hi);
            ap[2*v]   = ldsP[w][r][kb];
            ap[2*v+1] = ldsP[w][r][kb + 1];
        }
        #pragma unroll
        for (int j = 0; j < 8; ++j) {
            v16bf bv;
            #pragma unroll
            for (int v = 0; v < 8; ++v) {
                const int kb = kpack(v, hi);
                bv[2*v]   = ldsV[cur][kb][j * 16 + r];
                bv[2*v+1] = ldsV[cur][kb + 1][j * 16 + r];
            }
            o[j] = wmma_bf16(ap, bv, o[j]);
        }

        WAIT_ASYNC0();
        __syncthreads();
        cur ^= 1;
    }

    // ---- epilogue: reduce denominators once, normalize, scatter ----
    float linv[8];
    #pragma unroll
    for (int i = 0; i < 8; ++i) linv[i] = 1.0f / rowsum16(rowl[i]);

    #pragma unroll
    for (int j = 0; j < 8; ++j)
        #pragma unroll
        for (int i = 0; i < 8; ++i) {
            const int t = q0 + hi * 8 + i;
            const int d = j * 16 + r;
            const float val = o[j][i] * linv[i];
            y[((size_t)b * SEQ + t) * DIMC + (size_t)h * DH + d] = (bf16_t)val;
        }
}

// ---------------------------------------------------------------------------
extern "C" void kernel_launch(void* const* d_in, const int* in_sizes, int n_in,
                              void* d_out, int out_size, void* d_ws, size_t ws_size,
                              hipStream_t stream) {
    const float* x      = (const float*)d_in[0];
    const float* W_attn = (const float*)d_in[1];
    const float* b_attn = (const float*)d_in[2];
    const float* W_proj = (const float*)d_in[3];
    const float* b_proj = (const float*)d_in[4];
    float* out = (float*)d_out;

    char* ws = (char*)d_ws;
    bf16_t* xb  = (bf16_t*)(ws);                       //  33,554,432 B
    bf16_t* Wab = (bf16_t*)(ws + 33554432ull);         //  25,165,824 B
    bf16_t* Wpb = (bf16_t*)(ws + 58720256ull);         //   8,388,608 B
    bf16_t* qkv = (bf16_t*)(ws + 67108864ull);         // 100,663,296 B
    bf16_t* yb  = (bf16_t*)(ws + 167772160ull);        //  33,554,432 B
    // total workspace: 201,326,592 B

    // fp32 -> bf16 (each thread converts 4 elems)
    cvt_f32_bf16<<<MTOT * DIMC / 1024, 256, 0, stream>>>(x, xb);
    cvt_f32_bf16<<<DIMC * C3 / 1024, 256, 0, stream>>>(W_attn, Wab);
    cvt_f32_bf16<<<DIMC * DIMC / 1024, 256, 0, stream>>>(W_proj, Wpb);

    // qkv = x @ W_attn + b_attn   [8192 x 6144]
    gemm_bf16_wmma<true><<<dim3(MTOT / 128, C3 / 128), 256, 0, stream>>>(
        xb, Wab, b_attn, qkv, MTOT, C3, DIMC);

    // y = softmax(q k^T / sqrt(Dh)) v   (per b,h)
    attn_wmma<<<BATCH * NHEAD * (SEQ / 128), 256, 0, stream>>>(qkv, yb);

    // out = y @ W_proj + b_proj   [8192 x 2048], fp32 output
    gemm_bf16_wmma<false><<<dim3(MTOT / 128, DIMC / 128), 256, 0, stream>>>(
        yb, Wpb, b_proj, out, MTOT, DIMC, DIMC);
}
